// HungarianMatcher_4784593568282
// MI455X (gfx1250) — matrix operand order repaired and checked by reading, not compiled, not run on hardware
//
#include <hip/hip_runtime.h>
#include <hip/hip_bf16.h>

#define B_ 64
#define Q_ 900
#define T_ 300
#define C_ 92

typedef __attribute__((ext_vector_type(16))) __bf16 v16bf;
typedef __attribute__((ext_vector_type(8)))  __bf16 v8bf;
typedef __attribute__((ext_vector_type(8)))  float  v8f;

__device__ __forceinline__ unsigned short f2bf(float f) {
  unsigned int u = __float_as_uint(f);
  u += 0x7FFFu + ((u >> 16) & 1u);          // round-to-nearest-even
  return (unsigned short)(u >> 16);
}

__global__ __launch_bounds__(128)
void detr_cost_kernel(const float* __restrict__ pred_logits,
                      const float* __restrict__ pred_boxes,
                      const int*   __restrict__ tgt_labels,
                      const float* __restrict__ tgt_boxes,
                      float* __restrict__ out)
{
  constexpr int QTILES = (Q_ + 15) / 16;   // 57
  constexpr int TTILES = (T_ + 15) / 16;   // 19

  const int b  = blockIdx.x / QTILES;
  const int qt = blockIdx.x % QTILES;
  const int q0 = qt * 16;

  const int tid  = threadIdx.x;
  const int wave = tid >> 5;
  const int lane = tid & 31;

  __shared__ __align__(16) unsigned short sA[16 * 96]; // bf16 softmax rows, padded 92->96
  __shared__ float sPB[16 * 4];                        // pred boxes (cxcywh)
  __shared__ int   sLab[304];                          // labels, pad = -1
  __shared__ float sTB[304 * 4];                       // tgt boxes (cxcywh)

  // ---- stage labels / target boxes / pred boxes ----
  for (int t = tid; t < 304; t += 128) {
    const bool v = t < T_;
    sLab[t] = v ? tgt_labels[b * T_ + t] : -1;
    const float* tb = tgt_boxes + ((size_t)b * T_ + t) * 4;
    sTB[t * 4 + 0] = v ? tb[0] : 0.f;
    sTB[t * 4 + 1] = v ? tb[1] : 0.f;
    sTB[t * 4 + 2] = v ? tb[2] : 0.f;
    sTB[t * 4 + 3] = v ? tb[3] : 0.f;
  }
  for (int i = tid; i < 64; i += 128) {
    const int q = q0 + (i >> 2);
    sPB[i] = (q < Q_) ? pred_boxes[((size_t)b * Q_ + q) * 4 + (i & 3)] : 0.f;
  }

  // ---- fused softmax -> bf16 into LDS (one wave per query row) ----
  for (int rr = wave; rr < 16; rr += 4) {
    const int q = q0 + rr;
    if (q < Q_) {
      const float* src = pred_logits + ((size_t)b * Q_ + q) * C_;
      float x0 = src[lane];
      float x1 = src[lane + 32];
      float x2 = (lane < C_ - 64) ? src[lane + 64] : -3.0e38f;
      float mx = fmaxf(x0, fmaxf(x1, x2));
      #pragma unroll
      for (int o = 16; o > 0; o >>= 1) mx = fmaxf(mx, __shfl_xor(mx, o, 32));
      float e0 = __expf(x0 - mx);
      float e1 = __expf(x1 - mx);
      float e2 = (lane < C_ - 64) ? __expf(x2 - mx) : 0.f;
      float s = e0 + e1 + e2;
      #pragma unroll
      for (int o = 16; o > 0; o >>= 1) s += __shfl_xor(s, o, 32);
      const float inv = __builtin_amdgcn_rcpf(s);
      sA[rr * 96 + lane]      = f2bf(e0 * inv);
      sA[rr * 96 + lane + 32] = f2bf(e1 * inv);
      sA[rr * 96 + lane + 64] = (lane < C_ - 64) ? f2bf(e2 * inv) : (unsigned short)0;
    } else {
      sA[rr * 96 + lane]      = 0;
      sA[rr * 96 + lane + 32] = 0;
      sA[rr * 96 + lane + 64] = 0;
    }
  }
  __syncthreads();

  // ---- build A fragments (16x32 bf16 WMMA layout), 3 K-chunks covering K=96 ----
  const int m  = lane & 15;   // M row (A) / N col (B,D)
  const int hi = lane >> 4;
  v16bf a[3];
  #pragma unroll
  for (int c = 0; c < 3; ++c) {
    const int kk = c * 32;
    v8bf lo = *(const v8bf*)(&sA[m * 96 + kk + hi * 8]);        // K = kk+{0..7} / kk+{8..15}
    v8bf h8 = *(const v8bf*)(&sA[m * 96 + kk + 16 + hi * 8]);   // K = kk+{16..23} / kk+{24..31}
    #pragma unroll
    for (int k = 0; k < 8; ++k) { a[c][k] = lo[k]; a[c][k + 8] = h8[k]; }
  }

  const __bf16 onec  = (__bf16)1.0f;
  const __bf16 zeroc = (__bf16)0.0f;
  const int qbase = hi * 8;

  for (int tt = wave; tt < TTILES; tt += 4) {   // wave-uniform bound: EXEC all-ones at WMMA
    const int t0   = tt * 16;
    const int tcol = t0 + m;
    const int lab  = sLab[tcol];

    // cost_class via WMMA: prob (16x96 bf16) x one-hot(labels) (96x16 bf16)
    v8f acc = {0.f, 0.f, 0.f, 0.f, 0.f, 0.f, 0.f, 0.f};
    #pragma unroll
    for (int c = 0; c < 3; ++c) {
      const int base = c * 32 + hi * 8;    // B layout: lanes 0-15 K=0..7,16..23, lanes 16-31 K=8..15,24..31
      v16bf ob;
      #pragma unroll
      for (int k = 0; k < 16; ++k) ob[k] = (lab == base + k + (k >= 8 ? 8 : 0)) ? onec : zeroc;
      acc = __builtin_amdgcn_wmma_f32_16x16x32_bf16(false, a[c], false, ob,
                                                    (short)0, acc, false, false);
    }

    // target box (fixed per lane)
    const float tcx = sTB[tcol * 4 + 0], tcy = sTB[tcol * 4 + 1];
    const float tw  = sTB[tcol * 4 + 2], th  = sTB[tcol * 4 + 3];
    const float tx0 = tcx - 0.5f * tw, ty0 = tcy - 0.5f * th;
    const float tx1 = tcx + 0.5f * tw, ty1 = tcy + 0.5f * th;
    const float areat = fmaxf(tw, 0.f) * fmaxf(th, 0.f);

    #pragma unroll
    for (int r = 0; r < 8; ++r) {
      const int qr = qbase + r;            // D layout: lanes 0-15 -> M=r, lanes 16-31 -> M=8+r
      const int q  = q0 + qr;
      const float pcx = sPB[qr * 4 + 0], pcy = sPB[qr * 4 + 1];
      const float pw  = sPB[qr * 4 + 2], ph  = sPB[qr * 4 + 3];
      const float px0 = pcx - 0.5f * pw, py0 = pcy - 0.5f * ph;
      const float px1 = pcx + 0.5f * pw, py1 = pcy + 0.5f * ph;

      const float l1 = fabsf(pcx - tcx) + fabsf(pcy - tcy) +
                       fabsf(pw  - tw ) + fabsf(ph  - th );

      const float iw    = fmaxf(fminf(px1, tx1) - fmaxf(px0, tx0), 0.f);
      const float ih    = fmaxf(fminf(py1, ty1) - fmaxf(py0, ty0), 0.f);
      const float inter = iw * ih;
      const float areap = fmaxf(pw, 0.f) * fmaxf(ph, 0.f);
      const float uni   = areap + areat - inter;
      const float iou   = inter * __builtin_amdgcn_rcpf(uni);
      const float cw    = fmaxf(fmaxf(px1, tx1) - fminf(px0, tx0), 0.f);
      const float ch    = fmaxf(fmaxf(py1, ty1) - fminf(py0, ty0), 0.f);
      const float cvx   = cw * ch;
      const float giou  = iou - (cvx - uni) * __builtin_amdgcn_rcpf(cvx);

      // total = cost_bbox + cost_class + cost_giou = l1 - prob - giou
      const float cost = l1 - acc[r] - giou;
      if (q < Q_ && tcol < T_)
        out[((size_t)b * Q_ + q) * T_ + tcol] = cost;
    }
  }
}

extern "C" void kernel_launch(void* const* d_in, const int* in_sizes, int n_in,
                              void* d_out, int out_size, void* d_ws, size_t ws_size,
                              hipStream_t stream) {
  const float* pred_logits = (const float*)d_in[0];
  const float* pred_boxes  = (const float*)d_in[1];
  const int*   tgt_labels  = (const int*)d_in[2];
  const float* tgt_boxes   = (const float*)d_in[3];
  float* out = (float*)d_out;

  constexpr int QTILES = (Q_ + 15) / 16;   // 57
  dim3 grid(B_ * QTILES);                  // 3648 workgroups
  detr_cost_kernel<<<grid, 128, 0, stream>>>(pred_logits, pred_boxes,
                                             tgt_labels, tgt_boxes, out);
}
